// SupConLossV2_61942018343731
// MI455X (gfx1250) — matrix operand order repaired and checked by reading, not compiled, not run on hardware
//
#include <hip/hip_runtime.h>
#include <hip/hip_bf16.h>
#include <hip/hip_fp16.h>

// Problem constants (from the reference)
#define NSAMP 512
#define LDIM  1024
#define CDIM  128
#define KSEG  16
#define SEGD  8192            // (LDIM/KSEG)*CDIM
#define LC    (LDIM*CDIM)     // 131072 elements per sample
#define NN    (NSAMP*NSAMP)   // 262144 pairs
#define INV_TEMP (1.0f/0.7f)
#define NSUPER 16             // 512 / 32 supertiles per dim
#define NTRI   136            // NSUPER*(NSUPER+1)/2

typedef __attribute__((ext_vector_type(4)))  float    v4f;
typedef __attribute__((ext_vector_type(8)))  float    v8f;
typedef __attribute__((ext_vector_type(8)))  _Float16 v8h;
typedef __attribute__((ext_vector_type(16))) _Float16 v16h;

// ---------------------------------------------------------------------------
// Kernel 1: f32 -> f16 conversion, 8 elems/thread (2x b128 in, 1x b128 out)
// 67,108,864 elements = 32768 blocks x 256 threads x 8
// ---------------------------------------------------------------------------
__global__ void __launch_bounds__(256)
cvt_f32_to_f16(const float* __restrict__ in, _Float16* __restrict__ out) {
    long i = ((long)blockIdx.x * 256 + threadIdx.x) * 8;
    v4f a = *(const v4f*)(in + i);
    v4f b = *(const v4f*)(in + i + 4);
    v8h h;
    h[0] = (_Float16)a[0]; h[1] = (_Float16)a[1];
    h[2] = (_Float16)a[2]; h[3] = (_Float16)a[3];
    h[4] = (_Float16)b[0]; h[5] = (_Float16)b[1];
    h[6] = (_Float16)b[2]; h[7] = (_Float16)b[3];
    *(v8h*)(out + i) = h;
}

// ---------------------------------------------------------------------------
// WMMA helpers
// ---------------------------------------------------------------------------
__device__ __forceinline__ v16h load_afrag(const _Float16* p, int hs) {
    // A 16x32 f16, contiguous k order used here (B same): lanes<16 hold K 0..15, lanes>=16 K 16..31
    return *(const v16h*)(p + hs * 16);
}

__device__ __forceinline__ v8f wmma_f16(v16h a, v16h b, v8f c) {
    return __builtin_amdgcn_wmma_f32_16x16x32_f16(
               false, a, false, b, (short)0, c, false, false);
}

// normal store: value(v,lane) -> sims[k][a0+8*hs+v][b0+r]   (coalesced b32s)
__device__ __forceinline__ void store_tile(float* __restrict__ sims, int k,
                                           int a0, int b0, v8f acc,
                                           int hs, int r) {
    float* out = sims + (size_t)k * NN + (size_t)(a0 + hs * 8) * NSAMP + b0 + r;
    #pragma unroll
    for (int v = 0; v < 8; ++v) out[(size_t)v * NSAMP] = acc[v] * INV_TEMP;
}

// mirrored store: value(v,lane) -> sims[k][b0+r][a0+8*hs+v]  (2x b128/lane)
__device__ __forceinline__ void store_tile_T(float* __restrict__ sims, int k,
                                             int a0, int b0, v8f acc,
                                             int hs, int r) {
    float* out = sims + (size_t)k * NN + (size_t)(b0 + r) * NSAMP + a0 + 8 * hs;
    v4f lo, hi;
    lo[0] = acc[0] * INV_TEMP; lo[1] = acc[1] * INV_TEMP;
    lo[2] = acc[2] * INV_TEMP; lo[3] = acc[3] * INV_TEMP;
    hi[0] = acc[4] * INV_TEMP; hi[1] = acc[5] * INV_TEMP;
    hi[2] = acc[6] * INV_TEMP; hi[3] = acc[7] * INV_TEMP;
    *(v4f*)out       = lo;
    *(v4f*)(out + 4) = hi;
}

// ---------------------------------------------------------------------------
// Kernel 2: symmetric batched GEMM sims[k] = F_k * F_k^T / TEMP.
// One wave per 32x32 supertile (2x2 register-blocked 16x16 WMMA tiles),
// only i<=j supertiles computed (136 of 256); off-diagonal mirrored via
// transposed b128 stores. grid = (136, 16 segments), block = 1 wave.
// Per 32-K step: 4 fragment loads feed 4 v_wmma -> 2x intensity vs 1x1.
// ---------------------------------------------------------------------------
__global__ void __launch_bounds__(32)
seg_gemm_wmma(const _Float16* __restrict__ F, float* __restrict__ sims) {
    // decode triangular supertile index -> (i, j), i <= j
    int p = blockIdx.x, i = 0, rem = NSUPER;
    while (p >= rem) { p -= rem; rem--; i++; }
    const int j = i + p;
    const int k = blockIdx.y;

    const int lane = threadIdx.x;       // wave32
    const int hs   = lane >> 4;
    const int r    = lane & 15;

    const _Float16* A0 = F + (size_t)(i * 32 + r) * LC + (size_t)k * SEGD;
    const _Float16* A1 = A0 + (size_t)16 * LC;
    const _Float16* B0 = F + (size_t)(j * 32 + r) * LC + (size_t)k * SEGD;
    const _Float16* B1 = B0 + (size_t)16 * LC;

    v8f acc00 = {}, acc01 = {}, acc10 = {}, acc11 = {};
    #pragma unroll 2
    for (int kk = 0; kk < SEGD; kk += 32) {
        v16h a0 = load_afrag(A0 + kk, hs);
        v16h a1 = load_afrag(A1 + kk, hs);
        v16h b0 = *(const v16h*)(B0 + kk + hs * 16);
        v16h b1 = *(const v16h*)(B1 + kk + hs * 16);
        acc00 = wmma_f16(a0, b0, acc00);
        acc01 = wmma_f16(a0, b1, acc01);
        acc10 = wmma_f16(a1, b0, acc10);
        acc11 = wmma_f16(a1, b1, acc11);
    }

    const int ar = i * 32, bc = j * 32;
    store_tile(sims, k, ar,      bc,      acc00, hs, r);
    store_tile(sims, k, ar,      bc + 16, acc01, hs, r);
    store_tile(sims, k, ar + 16, bc,      acc10, hs, r);
    store_tile(sims, k, ar + 16, bc + 16, acc11, hs, r);
    if (i != j) {   // mirror into supertile (j, i)
        store_tile_T(sims, k, ar,      bc,      acc00, hs, r);
        store_tile_T(sims, k, ar,      bc + 16, acc01, hs, r);
        store_tile_T(sims, k, ar + 16, bc,      acc10, hs, r);
        store_tile_T(sims, k, ar + 16, bc + 16, acc11, hs, r);
    }
}

// ---------------------------------------------------------------------------
// Kernel 3: per-pair loss. One thread per (a,b): max over K, exp, top-4
// insertion, -log(top/total) * mask, block-reduce to a partial sum.
// ---------------------------------------------------------------------------
__global__ void __launch_bounds__(256)
loss_partial(const float* __restrict__ sims, const int* __restrict__ labels,
             float* __restrict__ partials) {
    const int pair = blockIdx.x * 256 + threadIdx.x;   // 0..NN-1
    const int a = pair >> 9;            // /512
    const int b = pair & (NSAMP - 1);

    float s[KSEG];
    float m = -1e30f;
    #pragma unroll
    for (int k = 0; k < KSEG; ++k) {
        s[k] = sims[(size_t)k * NN + pair];
        m = fmaxf(m, s[k]);
    }

    float total = 0.0f;
    float t0 = 0.0f, t1 = 0.0f, t2 = 0.0f, t3 = 0.0f;  // top-4 (exp > 0)
    #pragma unroll
    for (int k = 0; k < KSEG; ++k) {
        float e = __expf(s[k] - m);
        total += e;
        if (e > t3) {
            t3 = e;
            if (t3 > t2) { float x = t2; t2 = t3; t3 = x; }
            if (t2 > t1) { float x = t1; t1 = t2; t2 = x; }
            if (t1 > t0) { float x = t0; t0 = t1; t1 = x; }
        }
    }
    float top  = t0 + t1 + t2 + t3;
    float msk  = ((a != b) && (labels[a] == labels[b])) ? 1.0f : 0.0f;
    float loss = -__logf(top / total) * msk;

    __shared__ float sh[256];
    sh[threadIdx.x] = loss;
    __syncthreads();
    #pragma unroll
    for (int off = 128; off > 0; off >>= 1) {
        if (threadIdx.x < off) sh[threadIdx.x] += sh[threadIdx.x + off];
        __syncthreads();
    }
    if (threadIdx.x == 0) partials[blockIdx.x] = sh[0];
}

// ---------------------------------------------------------------------------
// Kernel 4: final reduction of partials -> d_out[0] (plain overwrite, graph-safe)
// ---------------------------------------------------------------------------
__global__ void __launch_bounds__(256)
finalize(const float* __restrict__ partials, float* __restrict__ out, int np) {
    __shared__ float sh[256];
    float s = 0.0f;
    for (int i = threadIdx.x; i < np; i += 256) s += partials[i];
    sh[threadIdx.x] = s;
    __syncthreads();
    #pragma unroll
    for (int off = 128; off > 0; off >>= 1) {
        if (threadIdx.x < off) sh[threadIdx.x] += sh[threadIdx.x + off];
        __syncthreads();
    }
    if (threadIdx.x == 0) out[0] = sh[0] / (float)NN;
}

// ---------------------------------------------------------------------------
extern "C" void kernel_launch(void* const* d_in, const int* in_sizes, int n_in,
                              void* d_out, int out_size, void* d_ws, size_t ws_size,
                              hipStream_t stream) {
    const float* features = (const float*)d_in[0];   // [512,1024,128] f32
    const int*   labels   = (const int*)d_in[1];     // [512] int
    // d_in[2] = preds : unused by the reference
    float* out = (float*)d_out;

    char* ws = (char*)d_ws;
    _Float16* f16  = (_Float16*)ws;                               // 134 MB
    size_t    f16b = (size_t)NSAMP * LC * sizeof(_Float16);
    float*    sims = (float*)(ws + f16b);                         // 16.8 MB
    size_t    simb = (size_t)KSEG * NN * sizeof(float);
    float*    part = (float*)(ws + f16b + simb);                  // 4 KB

    // 1) convert features once (8 elems/thread)
    cvt_f32_to_f16<<<32768, 256, 0, stream>>>(features, f16);

    // 2) 16 segment GEMMs: triangular supertiles, 2x2 WMMA register blocking
    dim3 g(NTRI, KSEG);
    seg_gemm_wmma<<<g, 32, 0, stream>>>(f16, sims);

    // 3) per-pair loss + block partials (1024 blocks)
    loss_partial<<<NN / 256, 256, 0, stream>>>(sims, labels, part);

    // 4) final mean
    finalize<<<1, 256, 0, stream>>>(part, out, NN / 256);
}
